// GPT_50740743635122
// MI455X (gfx1250) — matrix operand order repaired and checked by reading, not compiled, not run on hardware
//
#include <hip/hip_runtime.h>
#include <math.h>

// ---------------------------------------------------------------- constants
#define LCOUNT 12
#define DMODEL 768
#define FHID   3072
#define VOCAB  50304
#define BB     4
#define TT     1024
#define NTOK   (BB*TT)     // 4096

typedef __bf16 bf16;
typedef __attribute__((ext_vector_type(16))) __bf16 v16bf;
typedef __attribute__((ext_vector_type(8)))  __bf16 v8bf;
typedef __attribute__((ext_vector_type(8)))  float  v8f;

// ---------------------------------------------------------------- helpers
static __device__ inline bf16 f2bf(float f) {
    union { float f; unsigned u; } cv; cv.f = f;
    unsigned r = cv.u + 0x7fffu + ((cv.u >> 16) & 1u);   // RNE
    union { unsigned short s; bf16 b; } o; o.s = (unsigned short)(r >> 16);
    return o.b;
}

// ---- CDNA5 async global->LDS copy (ASYNCcnt), per CDNA5 ISA.
// VDST = VGPR holding LDS byte address (flat LDS aperture truncates to addr[31:0]),
// VADDR = 64-bit global address, SADDR = off.
static __device__ inline void async_ld_b128(void* lds, const void* g) {
    unsigned l = (unsigned)(size_t)lds;
    asm volatile("global_load_async_to_lds_b128 %0, %1, off"
                 :: "v"(l), "v"(g) : "memory");
}
static __device__ inline void wait_async0() {
    asm volatile("s_wait_asynccnt 0x0" ::: "memory");
}

// A fragment (16x32, M x K) from LDS/row-major bf16: lanes 0-15 hold
// K=0..15, lanes 16-31 hold K=16..31 (matches frag_b).
static __device__ inline v16bf frag_a(const bf16* p, int stride, int row, int k, int lane) {
    int koff = (lane & 16) ? 16 : 0;
    v8bf lo = *(const v8bf*)(p + (long long)row * stride + k + koff);
    v8bf hi = *(const v8bf*)(p + (long long)row * stride + k + 8 + koff);
    v16bf r;
#pragma unroll
    for (int i = 0; i < 8; ++i) { r[i] = lo[i]; r[i + 8] = hi[i]; }
    return r;
}
// B fragment (32x16, K x N) = column n of B = row n of weight W(N,K):
// lanes 0-15 hold K=0..15 of column (lane&15); lanes 16-31 hold K=16..31.
static __device__ inline v16bf frag_b(const bf16* p, long long stride, int col, int k, int lane) {
    int khalf = (lane & 16) ? 16 : 0;
    return *(const v16bf*)(p + (long long)col * stride + k + khalf);
}

static __device__ inline float warpSum(float v) {
#pragma unroll
    for (int o = 16; o > 0; o >>= 1) v += __shfl_xor(v, o, 32);
    return v;
}
static __device__ inline float warpMax(float v) {
#pragma unroll
    for (int o = 16; o > 0; o >>= 1) v = fmaxf(v, __shfl_xor(v, o, 32));
    return v;
}
static __device__ inline float blockSum(float v, float* red) {
    int lane = threadIdx.x & 31, w = threadIdx.x >> 5;
    v = warpSum(v);
    __syncthreads();
    if (lane == 0) red[w] = v;
    __syncthreads();
    float s = red[0];
#pragma unroll
    for (int i = 1; i < 8; ++i) s += red[i];
    return s;
}
static __device__ inline float blockMax(float v, float* red) {
    int lane = threadIdx.x & 31, w = threadIdx.x >> 5;
    v = warpMax(v);
    __syncthreads();
    if (lane == 0) red[w] = v;
    __syncthreads();
    float s = red[0];
#pragma unroll
    for (int i = 1; i < 8; ++i) s = fmaxf(s, red[i]);
    return s;
}

// ---------------------------------------------------------------- GEMM
// C(M,N) = A(M,K,row-major bf16) * W(N,K,row-major bf16)^T   [+ epilogue]
// 256 threads (8 wave32); C tile 128x128; wave tile 32x64 (2x4 WMMA accum).
// Tiles staged with double-buffered async global->LDS copies: one barrier per
// K step; WMMA on buffer `cur` overlaps the async fill of buffer `cur^1`.
static constexpr int EPI_F32_BIAS  = 0;  // fp32 out, + bias[col]
static constexpr int EPI_BF16      = 1;  // bf16 out
static constexpr int EPI_BF16_VT   = 2;  // bf16 out, transposed per batch of TT rows
static constexpr int EPI_SCORES    = 3;  // fp32 out = acc*scale - |row-col|
static constexpr int EPI_GELU_BF16 = 4;  // bf16 out = gelu(acc + bias[col])

template<int EPI>
__global__ __launch_bounds__(256)
void gemm_bf16(const bf16* __restrict__ A, const bf16* __restrict__ Wt,
               void* __restrict__ Cout, const float* __restrict__ bias,
               int K, int lda, int ldb, int ldc,
               long long strideA, long long strideB, long long strideC,
               float scale)
{
    __shared__ bf16 As[2][128 * 32];
    __shared__ bf16 Bs[2][128 * 32];
    const int tid  = threadIdx.x;
    const int lane = tid & 31, wave = tid >> 5;
    const int wm = wave & 3, wn = wave >> 2;      // 4 (M) x 2 (N) waves
    const int row0 = blockIdx.y * 128;
    const int col0 = blockIdx.x * 128;
    const int b    = blockIdx.z;

    const int rL = tid >> 1;            // 0..127 (tile row)
    const int hL = (tid & 1) * 16;      // K half within 32-wide tile
    const bf16* ga = A  + (long long)b * strideA + (long long)(row0 + rL) * lda + hL;
    const bf16* gb = Wt + (long long)b * strideB + (long long)(col0 + rL) * ldb + hL;
    bf16* sa0 = &As[0][rL * 32 + hL];
    bf16* sb0 = &Bs[0][rL * 32 + hL];
    bf16* sa1 = &As[1][rL * 32 + hL];
    bf16* sb1 = &Bs[1][rL * 32 + hL];

    v8f acc[2][4];
#pragma unroll
    for (int i = 0; i < 2; ++i)
#pragma unroll
        for (int j = 0; j < 4; ++j)
#pragma unroll
            for (int e = 0; e < 8; ++e) acc[i][j][e] = 0.f;

    // prologue: async-fill buffer 0 with K tile 0
    async_ld_b128(sa0,     ga);
    async_ld_b128(sa0 + 8, ga + 8);
    async_ld_b128(sb0,     gb);
    async_ld_b128(sb0 + 8, gb + 8);

    int cur = 0;
    for (int k0 = 0; k0 < K; k0 += 32) {
        wait_async0();          // this wave's LDS writes landed
        __syncthreads();        // -> every wave's writes visible; prev reads done
        if (k0 + 32 < K) {      // async-fill the other buffer with the next tile
            bf16* na = cur ? sa0 : sa1;
            bf16* nb = cur ? sb0 : sb1;
            async_ld_b128(na,     ga + k0 + 32);
            async_ld_b128(na + 8, ga + k0 + 40);
            async_ld_b128(nb,     gb + k0 + 32);
            async_ld_b128(nb + 8, gb + k0 + 40);
        }
        const bf16* at = As[cur];
        const bf16* bt = Bs[cur];
        v16bf af[2], bfv[4];
#pragma unroll
        for (int i = 0; i < 2; ++i) af[i]  = frag_a(at, 32, wm * 32 + i * 16 + (lane & 15), 0, lane);
#pragma unroll
        for (int j = 0; j < 4; ++j) bfv[j] = frag_b(bt, 32, wn * 64 + j * 16 + (lane & 15), 0, lane);
#pragma unroll
        for (int i = 0; i < 2; ++i)
#pragma unroll
            for (int j = 0; j < 4; ++j)
                acc[i][j] = __builtin_amdgcn_wmma_f32_16x16x32_bf16(
                    false, af[i], false, bfv[j], (short)0, acc[i][j], false, false);
        cur ^= 1;
    }

    // epilogue: C layout -> col = lane&15, row = vgpr + 8*(lane>=16)
    const int hi8 = (lane & 16) ? 8 : 0;
    const int cn  = lane & 15;
#pragma unroll
    for (int i = 0; i < 2; ++i) {
#pragma unroll
        for (int j = 0; j < 4; ++j) {
#pragma unroll
            for (int v = 0; v < 8; ++v) {
                int row = row0 + wm * 32 + i * 16 + v + hi8;
                int col = col0 + wn * 64 + j * 16 + cn;
                float val = acc[i][j][v];
                if (EPI == EPI_F32_BIAS) {
                    ((float*)Cout)[(long long)b * strideC + (long long)row * ldc + col] = val + bias[col];
                } else if (EPI == EPI_BF16) {
                    ((bf16*)Cout)[(long long)b * strideC + (long long)row * ldc + col] = f2bf(val);
                } else if (EPI == EPI_BF16_VT) {
                    long long bb = row >> 10, s = row & (TT - 1);   // V^T: [b][col][s]
                    ((bf16*)Cout)[bb * (long long)(DMODEL * TT) + (long long)col * TT + s] = f2bf(val);
                } else if (EPI == EPI_SCORES) {
                    float pb = -fabsf((float)(row - col));          // ALiBi-style bias
                    ((float*)Cout)[(long long)b * strideC + (long long)row * ldc + col] = val * scale + pb;
                } else if (EPI == EPI_GELU_BF16) {
                    float xv = val + bias[col];
                    float g = 0.5f * xv * (1.f + erff(xv * 0.70710678118654752f));
                    ((bf16*)Cout)[(long long)b * strideC + (long long)row * ldc + col] = f2bf(g);
                }
            }
        }
    }
}

// ---------------------------------------------------------------- elementwise
__global__ void k_embed(const int* __restrict__ idx, const float* __restrict__ wte,
                        float* __restrict__ x) {
    int row = blockIdx.x;
    long long tok = idx[row];
    const float* src = wte + tok * DMODEL;
    float* dst = x + (long long)row * DMODEL;
    for (int c = threadIdx.x; c < DMODEL; c += blockDim.x) dst[c] = src[c];
}

__global__ void k_f32_to_bf16(const float* __restrict__ in, bf16* __restrict__ out, long long n) {
    for (long long i = (long long)blockIdx.x * blockDim.x + threadIdx.x; i < n;
         i += (long long)gridDim.x * blockDim.x)
        out[i] = f2bf(in[i]);
}

// xl = x + pos_embed[l]; prior = dyt3(xl); h = dyt(xl) (bf16)
__global__ void k_pre(const float* __restrict__ x, const float* __restrict__ pe,
                      const float* __restrict__ alpha, const float* __restrict__ w,
                      const float* __restrict__ bsrc, int l,
                      float* __restrict__ prior, bf16* __restrict__ h) {
    float a0 = alpha[l * 3 + 0], a2 = alpha[l * 3 + 2];
    const float* w0 = w + (long long)(l * 3 + 0) * DMODEL;
    const float* w2 = w + (long long)(l * 3 + 2) * DMODEL;
    const float* b0 = bsrc + (long long)(l * 3 + 0) * DMODEL;
    const float* b2 = bsrc + (long long)(l * 3 + 2) * DMODEL;
    const float* pel = pe + (long long)l * DMODEL;
    for (long long i = (long long)blockIdx.x * blockDim.x + threadIdx.x;
         i < (long long)NTOK * DMODEL; i += (long long)gridDim.x * blockDim.x) {
        int d = (int)(i % DMODEL);
        float xl = x[i] + pel[d];
        prior[i] = tanhf(a2 * xl) * w2[d] + b2[d];
        h[i] = f2bf(tanhf(a0 * xl) * w0[d] + b0[d]);
    }
}

__global__ void k_dyt2(const float* __restrict__ ao, const float* __restrict__ alpha,
                       const float* __restrict__ w, const float* __restrict__ bsrc,
                       int l, bf16* __restrict__ h2) {
    float a1 = alpha[l * 3 + 1];
    const float* w1 = w + (long long)(l * 3 + 1) * DMODEL;
    const float* b1 = bsrc + (long long)(l * 3 + 1) * DMODEL;
    for (long long i = (long long)blockIdx.x * blockDim.x + threadIdx.x;
         i < (long long)NTOK * DMODEL; i += (long long)gridDim.x * blockDim.x) {
        int d = (int)(i % DMODEL);
        h2[i] = f2bf(tanhf(a1 * ao[i]) * w1[d] + b1[d]);
    }
}

// softmax over TT columns -> bf16 attention weights (one row per block)
__global__ __launch_bounds__(256) void k_softmax_rows(const float* __restrict__ s,
                                                      bf16* __restrict__ out) {
    __shared__ float red[8];
    const float* row = s + (long long)blockIdx.x * TT;
    bf16* orow = out + (long long)blockIdx.x * TT;
    float v[4], mx = -3.4e38f;
#pragma unroll
    for (int j = 0; j < 4; ++j) { v[j] = row[threadIdx.x + j * 256]; mx = fmaxf(mx, v[j]); }
    mx = blockMax(mx, red);
    float sm = 0.f;
#pragma unroll
    for (int j = 0; j < 4; ++j) { v[j] = __expf(v[j] - mx); sm += v[j]; }
    sm = blockSum(sm, red);
    float inv = 1.f / sm;
#pragma unroll
    for (int j = 0; j < 4; ++j) orow[threadIdx.x + j * 256] = f2bf(v[j] * inv);
}

// posterior = mlp + prior -> x ; KL row term (one row of D per block)
__global__ __launch_bounds__(256) void k_post_kl(const float* __restrict__ mlp,
                                                 const float* __restrict__ prior,
                                                 float* __restrict__ x,
                                                 float* __restrict__ klacc) {
    __shared__ float red[8];
    long long base = (long long)blockIdx.x * DMODEL;
    float pv[3], qv[3], mv[3];
    float mp = -3.4e38f, mq = -3.4e38f, mm = -3.4e38f;
#pragma unroll
    for (int j = 0; j < 3; ++j) {
        int d = threadIdx.x + j * 256;
        float pr = prior[base + d];
        float po = mlp[base + d] + pr;
        x[base + d] = po;
        pv[j] = pr; qv[j] = po; mv[j] = 0.5f * (pr + po);
        mp = fmaxf(mp, pr); mq = fmaxf(mq, po); mm = fmaxf(mm, mv[j]);
    }
    mp = blockMax(mp, red); mq = blockMax(mq, red); mm = blockMax(mm, red);
    float sp = 0.f, sq = 0.f, sm = 0.f;
#pragma unroll
    for (int j = 0; j < 3; ++j) {
        sp += __expf(pv[j] - mp); sq += __expf(qv[j] - mq); sm += __expf(mv[j] - mm);
    }
    sp = blockSum(sp, red); sq = blockSum(sq, red); sm = blockSum(sm, red);
    float lseP = mp + __logf(sp), lseQ = mq + __logf(sq), lseM = mm + __logf(sm);
    float t = 0.f;
#pragma unroll
    for (int j = 0; j < 3; ++j) {
        float lm = mv[j] - lseM;
        t += __expf(lm) * (2.f * lm - (pv[j] - lseP) - (qv[j] - lseQ));
    }
    t = blockSum(t, red);
    if (threadIdx.x == 0) atomicAdd(klacc, 0.5f * t);   // 0.5*(kl_p + kl_q)
}

// ---------------------------------------------------------------- lm head
// WG owns 16 token rows; streams V in 128-col WMMA tiles with online LSE.
// Writes full logit rows for last tokens into d_out; accumulates NLL sum.
__global__ __launch_bounds__(256)
void k_lm_head(const bf16* __restrict__ xb, const bf16* __restrict__ wteb,
               const int* __restrict__ targets, float* __restrict__ logits,
               float* __restrict__ acc) {
    __shared__ bf16  Arow[16 * DMODEL];     // 24KB resident activation slab
    __shared__ float cbuf[16 * 128];        // 8KB C tile
    __shared__ float pmax[16 * 8], psum[16 * 8];
    __shared__ float rmax[16], rsum[16], tl[16];
    const int tid = threadIdx.x, lane = tid & 31, wave = tid >> 5;
    const int r0 = blockIdx.x * 16;
    {   // async-fill the activation slab (6 x b128 per thread)
        const bf16* src = xb + (long long)r0 * DMODEL;
        for (int i = tid; i < 16 * DMODEL / 8; i += 256)
            async_ld_b128(&Arow[i * 8], src + i * 8);
    }
    if (tid < 16) { rmax[tid] = -3.4e38f; rsum[tid] = 0.f; tl[tid] = 0.f; }
    wait_async0();
    __syncthreads();

    const int colw  = wave * 16 + (lane & 15);
    const int khalf = (lane & 16) ? 16 : 0;
    for (int n0 = 0; n0 < VOCAB; n0 += 128) {
        v8f accv;
#pragma unroll
        for (int e = 0; e < 8; ++e) accv[e] = 0.f;
        const bf16* wrow = wteb + (long long)(n0 + colw) * DMODEL + khalf;
        for (int k = 0; k < DMODEL; k += 32) {
            v16bf a = frag_a(Arow, DMODEL, lane & 15, k, lane);
            v16bf bfv = *(const v16bf*)(wrow + k);      // L2-resident wte bf16
            accv = __builtin_amdgcn_wmma_f32_16x16x32_bf16(
                false, a, false, bfv, (short)0, accv, false, false);
        }
        const int hi8 = (lane & 16) ? 8 : 0;
#pragma unroll
        for (int v = 0; v < 8; ++v) cbuf[(v + hi8) * 128 + colw] = accv[v];
        __syncthreads();
        if (tid < 128) {
            int r = tid >> 3, seg = tid & 7;
            int grow = r0 + r;
            int tgt  = targets[grow];
            bool last = ((grow & (TT - 1)) == (TT - 1));
            int bi = grow >> 10;
            float vals[16], lmax = -3.4e38f;
#pragma unroll
            for (int c = 0; c < 16; ++c) {
                int cc = seg * 16 + c;
                float val = cbuf[r * 128 + cc];
                vals[c] = val;
                int n = n0 + cc;
                if (n == tgt) tl[r] = val;
                if (last) logits[(long long)bi * VOCAB + n] = val;
                lmax = fmaxf(lmax, val);
            }
            float ls = 0.f;
#pragma unroll
            for (int c = 0; c < 16; ++c) ls += __expf(vals[c] - lmax);
            pmax[r * 8 + seg] = lmax; psum[r * 8 + seg] = ls;
        }
        __syncthreads();
        if (tid < 16) {
            float m = rmax[tid], s = rsum[tid];
#pragma unroll
            for (int seg = 0; seg < 8; ++seg) {
                float pm = pmax[tid * 8 + seg], ps = psum[tid * 8 + seg];
                float nm = fmaxf(m, pm);
                s = s * __expf(m - nm) + ps * __expf(pm - nm);
                m = nm;
            }
            rmax[tid] = m; rsum[tid] = s;
        }
        __syncthreads();
    }
    if (tid < 16) {
        float nll = (rmax[tid] + __logf(rsum[tid])) - tl[tid];
        atomicAdd(&acc[0], nll);
    }
}

__global__ void k_init_acc(float* a) { a[0] = 0.f; a[1] = 0.f; }
__global__ void k_finalize(const float* __restrict__ acc, float* __restrict__ out) {
    out[0] = acc[0] / (float)NTOK + 0.1f * (acc[1] / (float)BB);
}

// ---------------------------------------------------------------- launch
extern "C" void kernel_launch(void* const* d_in, const int* in_sizes, int n_in,
                              void* d_out, int out_size, void* d_ws, size_t ws_size,
                              hipStream_t stream) {
    const int*   idx  = (const int*)d_in[0];
    const int*   tgt  = (const int*)d_in[1];
    const float* wte  = (const float*)d_in[2];
    const float* pe   = (const float*)d_in[3];
    const float* dya  = (const float*)d_in[4];
    const float* dyw  = (const float*)d_in[5];
    const float* dyb  = (const float*)d_in[6];
    const float* Wq   = (const float*)d_in[7];
    const float* Wk   = (const float*)d_in[8];
    const float* Wv   = (const float*)d_in[9];
    const float* Wp   = (const float*)d_in[10];
    const float* bp   = (const float*)d_in[11];
    const float* We   = (const float*)d_in[12];
    const float* be   = (const float*)d_in[13];
    const float* Wc   = (const float*)d_in[14];
    const float* bc   = (const float*)d_in[15];
    float* out = (float*)d_out;

    char* ws = (char*)d_ws;
    size_t off = 0;
    auto alloc = [&](size_t bytes) -> char* {
        char* p = ws + off; off += (bytes + 255) & ~(size_t)255; return p;
    };
    float* xf    = (float*)alloc(sizeof(float) * (size_t)NTOK * DMODEL);
    float* prior = (float*)alloc(sizeof(float) * (size_t)NTOK * DMODEL);
    bf16*  hb    = (bf16*) alloc(2ull * NTOK * DMODEL);
    bf16*  qb    = (bf16*) alloc(2ull * NTOK * DMODEL);
    bf16*  kb    = (bf16*) alloc(2ull * NTOK * DMODEL);
    bf16*  vtb   = (bf16*) alloc(2ull * NTOK * DMODEL);      // V^T per batch
    float* sc    = (float*)alloc(sizeof(float) * (size_t)BB * TT * TT);
    bf16*  awb   = (bf16*) alloc(2ull * BB * TT * TT);
    bf16*  aob   = (bf16*) alloc(2ull * NTOK * DMODEL);
    float* aopf  = (float*)alloc(sizeof(float) * (size_t)NTOK * DMODEL);
    bf16*  h2b   = (bf16*) alloc(2ull * NTOK * DMODEL);
    bf16*  hidb  = (bf16*) alloc(2ull * NTOK * FHID);
    float* mlpf  = (float*)alloc(sizeof(float) * (size_t)NTOK * DMODEL);
    bf16*  wqb   = (bf16*) alloc(2ull * DMODEL * DMODEL);
    bf16*  wkb   = (bf16*) alloc(2ull * DMODEL * DMODEL);
    bf16*  wvb   = (bf16*) alloc(2ull * DMODEL * DMODEL);
    bf16*  wpb   = (bf16*) alloc(2ull * DMODEL * DMODEL);
    bf16*  web   = (bf16*) alloc(2ull * FHID * DMODEL);
    bf16*  wcb   = (bf16*) alloc(2ull * DMODEL * FHID);
    bf16*  wteb  = (bf16*) alloc(2ull * VOCAB * DMODEL);
    bf16*  xbb   = (bf16*) alloc(2ull * NTOK * DMODEL);
    float* acc   = (float*)alloc(2 * sizeof(float));

    k_init_acc<<<1, 1, 0, stream>>>(acc);
    k_embed<<<NTOK, 256, 0, stream>>>(idx, wte, xf);
    k_f32_to_bf16<<<2048, 256, 0, stream>>>(wte, wteb, (long long)VOCAB * DMODEL);

    const float scale = 1.0f / sqrtf((float)DMODEL);
    const dim3 blk(256);

    for (int l = 0; l < LCOUNT; ++l) {
        k_f32_to_bf16<<<512, 256, 0, stream>>>(Wq + (long long)l * DMODEL * DMODEL, wqb, (long long)DMODEL * DMODEL);
        k_f32_to_bf16<<<512, 256, 0, stream>>>(Wk + (long long)l * DMODEL * DMODEL, wkb, (long long)DMODEL * DMODEL);
        k_f32_to_bf16<<<512, 256, 0, stream>>>(Wv + (long long)l * DMODEL * DMODEL, wvb, (long long)DMODEL * DMODEL);
        k_f32_to_bf16<<<512, 256, 0, stream>>>(Wp + (long long)l * DMODEL * DMODEL, wpb, (long long)DMODEL * DMODEL);
        k_f32_to_bf16<<<1024, 256, 0, stream>>>(We + (long long)l * FHID * DMODEL, web, (long long)FHID * DMODEL);
        k_f32_to_bf16<<<1024, 256, 0, stream>>>(Wc + (long long)l * DMODEL * FHID, wcb, (long long)DMODEL * FHID);

        k_pre<<<2048, 256, 0, stream>>>(xf, pe, dya, dyw, dyb, l, prior, hb);

        // Q, K, V projections (M=4096, N=768, K=768)
        gemm_bf16<EPI_BF16><<<dim3(DMODEL / 128, NTOK / 128, 1), blk, 0, stream>>>(
            hb, wqb, qb, nullptr, DMODEL, DMODEL, DMODEL, DMODEL, 0, 0, 0, 1.f);
        gemm_bf16<EPI_BF16><<<dim3(DMODEL / 128, NTOK / 128, 1), blk, 0, stream>>>(
            hb, wkb, kb, nullptr, DMODEL, DMODEL, DMODEL, DMODEL, 0, 0, 0, 1.f);
        gemm_bf16<EPI_BF16_VT><<<dim3(DMODEL / 128, NTOK / 128, 1), blk, 0, stream>>>(
            hb, wvb, vtb, nullptr, DMODEL, DMODEL, DMODEL, DMODEL, 0, 0, 0, 1.f);

        // scores = Q K^T * scale + pbias  (batched, M=N=1024, K=768)
        gemm_bf16<EPI_SCORES><<<dim3(TT / 128, TT / 128, BB), blk, 0, stream>>>(
            qb, kb, sc, nullptr, DMODEL, DMODEL, DMODEL, TT,
            (long long)TT * DMODEL, (long long)TT * DMODEL, (long long)TT * TT, scale);
        k_softmax_rows<<<BB * TT, 256, 0, stream>>>(sc, awb);

        // ao = aw @ V   (A=aw 1024x1024, "W"=V^T 768x1024, batched)
        gemm_bf16<EPI_BF16><<<dim3(DMODEL / 128, TT / 128, BB), blk, 0, stream>>>(
            awb, vtb, aob, nullptr, TT, TT, TT, DMODEL,
            (long long)TT * TT, (long long)DMODEL * TT, (long long)TT * DMODEL, 1.f);

        // projection + bias (fp32 out)
        gemm_bf16<EPI_F32_BIAS><<<dim3(DMODEL / 128, NTOK / 128, 1), blk, 0, stream>>>(
            aob, wpb, aopf, bp + (long long)l * DMODEL, DMODEL, DMODEL, DMODEL, DMODEL, 0, 0, 0, 1.f);
        k_dyt2<<<2048, 256, 0, stream>>>(aopf, dya, dyw, dyb, l, h2b);

        // MLP: gelu(h2 @ We^T + be) @ Wc^T + bc
        gemm_bf16<EPI_GELU_BF16><<<dim3(FHID / 128, NTOK / 128, 1), blk, 0, stream>>>(
            h2b, web, hidb, be + (long long)l * FHID, DMODEL, DMODEL, DMODEL, FHID, 0, 0, 0, 1.f);
        gemm_bf16<EPI_F32_BIAS><<<dim3(DMODEL / 128, NTOK / 128, 1), blk, 0, stream>>>(
            hidb, wcb, mlpf, bc + (long long)l * DMODEL, FHID, FHID, FHID, DMODEL, 0, 0, 0, 1.f);

        // posterior -> x, accumulate KL
        k_post_kl<<<NTOK, 256, 0, stream>>>(mlpf, prior, xf, acc + 1);
    }

    k_f32_to_bf16<<<2048, 256, 0, stream>>>(xf, xbb, (long long)NTOK * DMODEL);
    k_lm_head<<<NTOK / 16, 256, 0, stream>>>(xbb, wteb, tgt, out, acc);
    k_finalize<<<1, 1, 0, stream>>>(acc, out + (long long)BB * VOCAB);

    (void)in_sizes; (void)n_in; (void)out_size; (void)ws_size;
}
